// Attention1_2714419331284
// MI455X (gfx1250) — matrix operand order repaired and checked by reading, not compiled, not run on hardware
//
#include <hip/hip_runtime.h>
#include <math.h>

typedef __attribute__((ext_vector_type(16))) __bf16 v16bf;
typedef __attribute__((ext_vector_type(8)))  float  v8f;
typedef __attribute__((ext_vector_type(4)))  unsigned int u32x4;
typedef __attribute__((ext_vector_type(8)))  int          i32x8;
typedef __attribute__((ext_vector_type(4)))  int          i32x4;

#if defined(__has_builtin)
#  if __has_builtin(__builtin_amdgcn_tensor_load_to_lds) && \
      __has_builtin(__builtin_amdgcn_s_wait_tensorcnt)
#    define USE_TDM 1
#  endif
#endif
#ifndef USE_TDM
#  define USE_TDM 0
#endif

#define B_   8
#define N_   1024
#define C0_  256
#define KV_  512
#define H_   4

// ---------------------------------------------------------------------------
// fp32 -> bf16 elementwise conversion (grid-stride)
// ---------------------------------------------------------------------------
__global__ __launch_bounds__(256) void f32_to_bf16(const float* __restrict__ in,
                                                   __bf16* __restrict__ out,
                                                   size_t n) {
    size_t i = (size_t)blockIdx.x * blockDim.x + threadIdx.x;
    size_t stride = (size_t)gridDim.x * blockDim.x;
    for (; i < n; i += stride) out[i] = (__bf16)in[i];
}

// ---------------------------------------------------------------------------
// Generic bf16 WMMA GEMM:  C[z] = A[z] (MxK, row-major) * Bcm[z] (KxN, stored
// column-major as [N][K]).  Block = 256 threads = 8 wave32; macro tile 64x128,
// each wave owns a 16x64 strip (A-fragment reuse across 4 WMMAs / K-step).
// B panel (128 cols x 32 K, 8 KB) is staged into LDS by the Tensor Data Mover
// (double-buffered: DMA of panel k+32 overlaps WMMAs on panel k); fallback is
// a cooperative global->LDS copy when the TDM builtin is unavailable.
// outMode: 0 = bf16 row-major [M][N], 1 = bf16 transposed [N][M],
//          2 = fp32 row-major scaled.
// ---------------------------------------------------------------------------
__global__ __launch_bounds__(256) void wmma_gemm(
    const __bf16* __restrict__ A, const __bf16* __restrict__ Bm,
    void* __restrict__ Cout,
    int M, int N, int K,
    size_t aStride, int aShift,          // A batch index = z >> aShift
    size_t bStride, int bMask,           // B batch index = z & bMask
    size_t cStride,                      // C batch index = z
    int outMode, float scale)
{
    __shared__ __bf16 Bsh[2][128 * 32];  // double-buffered B panel, 16 KB

    const int tid = threadIdx.x;
    const int l   = tid & 31;
    const int w   = tid >> 5;
    const int m0  = blockIdx.y * 64 + (w >> 1) * 16;
    const int n0l = (w & 1) * 64;                    // column base within panel
    const int colBase = blockIdx.x * 128;            // panel base in B
    const int z   = blockIdx.z;

    const __bf16* Ab = A  + (size_t)(z >> aShift) * aStride;
    const __bf16* Bb = Bm + (size_t)(z & bMask)  * bStride;

    const int mA = m0 + (l & 15);             // A row for this lane
    const int kA = (l < 16) ? 0 : 16;         // A K sub-offset (contiguous k = 16h + s)
    const int kB = (l < 16) ? 0 : 16;         // B K sub-offset (contiguous k = 16h + s)
    const int nc = l & 15;                    // column-within-tile

    // ---- stage one 128x32 B panel into Bsh[buf] starting at K-offset kk ----
    auto issue_panel = [&](int kk, int buf) {
#if USE_TDM
        if (w == 0) {
            // 2D TDM descriptor: dim0 = K (contiguous, stride K), dim1 = cols.
            // tile_dim0 = 32 K-elems (64 B rows), tile_dim1 = 128 cols.
            unsigned long long ga = (unsigned long long)Bb +
                ((unsigned long long)(unsigned)colBase * (unsigned long long)(unsigned)K
                 + (unsigned long long)(unsigned)kk) * 2ull;
            unsigned lds = (unsigned)(unsigned long long)&Bsh[buf][0];
            u32x4 g0 = { 1u,                                   // count = 1 descriptor
                         lds,                                  // lds_addr [63:32]
                         (unsigned)ga,                         // global_addr [95:64]
                         (unsigned)(ga >> 32) | (2u << 30) };  // addr hi | type=2
            i32x8 g1 = { (int)(1u << 16),                      // data_size = 2 B
                         (int)(((unsigned)K & 0xFFFFu) << 16), // tensor_dim0 lo
                         (int)(((unsigned)K >> 16) |
                               (((unsigned)N & 0xFFFFu) << 16)), // dim0 hi | dim1 lo
                         (int)(((unsigned)N >> 16) | (32u << 16)), // dim1 hi | tile_dim0
                         128,                                  // tile_dim1 = 128 cols
                         K,                                    // tensor_dim0_stride lo
                         0, 0 };
            i32x4 zz = { 0, 0, 0, 0 };
#if __clang_major__ >= 23
            i32x8 z8 = { 0, 0, 0, 0, 0, 0, 0, 0 };
            __builtin_amdgcn_tensor_load_to_lds(g0, g1, zz, zz, z8, 0);
#else
            __builtin_amdgcn_tensor_load_to_lds(g0, g1, zz, zz, 0);
#endif
        }
#else
        // cooperative copy: 256 threads x 32 B = 8 KB panel
        const int col  = tid >> 1;
        const int half = tid & 1;
        v16bf s = *(const v16bf*)(Bb + (size_t)(colBase + col) * K +
                                  (size_t)(kk + half * 16));
        *(v16bf*)(&Bsh[buf][(size_t)col * 32 + half * 16]) = s;
#endif
    };

    v8f acc[4];
#pragma unroll
    for (int t = 0; t < 4; ++t) acc[t] = (v8f){0.f,0.f,0.f,0.f,0.f,0.f,0.f,0.f};

    // prologue: fetch panel 0
    issue_panel(0, 0);
#if USE_TDM
    __builtin_amdgcn_s_wait_tensorcnt(0);
#endif
    __syncthreads();

    for (int k0 = 0; k0 < K; k0 += 32) {
        const int cur = (k0 >> 5) & 1;
        if (k0 + 32 < K) issue_panel(k0 + 32, cur ^ 1);   // DMA next panel (overlap)

        // ---- A fragment: two contiguous 16B chunks per lane (row-major A) ----
        union { unsigned int u[8]; v16bf v; } af;
        const unsigned int* ap =
            (const unsigned int*)(Ab + (size_t)mA * K + (size_t)(k0 + kA));
#pragma unroll
        for (int vv = 0; vv < 4; ++vv) af.u[vv] = ap[vv];        // K = kA+0..7
#pragma unroll
        for (int vv = 4; vv < 8; ++vv) af.u[vv] = ap[vv];        // K = kA+8..15

        // ---- 4 B fragments from LDS (16 contiguous bf16 per lane) ----
        const __bf16* bp = &Bsh[cur][0];
        v16bf b0 = *(const v16bf*)(bp + (size_t)(n0l +  0 + nc) * 32 + kB);
        v16bf b1 = *(const v16bf*)(bp + (size_t)(n0l + 16 + nc) * 32 + kB);
        v16bf b2 = *(const v16bf*)(bp + (size_t)(n0l + 32 + nc) * 32 + kB);
        v16bf b3 = *(const v16bf*)(bp + (size_t)(n0l + 48 + nc) * 32 + kB);

        // same A operand back-to-back -> set reuse-A hint on WMMAs 2..4
        acc[0] = __builtin_amdgcn_wmma_f32_16x16x32_bf16(
            false, af.v, false, b0, (short)0, acc[0], false, false);
        acc[1] = __builtin_amdgcn_wmma_f32_16x16x32_bf16(
            false, af.v, false, b1, (short)0, acc[1], true, false);
        acc[2] = __builtin_amdgcn_wmma_f32_16x16x32_bf16(
            false, af.v, false, b2, (short)0, acc[2], true, false);
        acc[3] = __builtin_amdgcn_wmma_f32_16x16x32_bf16(
            false, af.v, false, b3, (short)0, acc[3], true, false);

#if USE_TDM
        __builtin_amdgcn_s_wait_tensorcnt(0);   // next panel landed (overlapped)
#endif
        __syncthreads();
    }

    // ---- epilogue: lane holds C[(l<16?v:8+v)][l&15] of each 16x16 tile ----
    const int n0 = colBase + n0l;
    if (outMode == 2) {
        float* O = (float*)Cout + (size_t)z * cStride;
#pragma unroll
        for (int t = 0; t < 4; ++t) {
            const int n = n0 + t * 16 + nc;
#pragma unroll
            for (int vv = 0; vv < 8; ++vv) {
                const int m = m0 + ((l < 16) ? vv : 8 + vv);
                O[(size_t)m * N + n] = acc[t][vv] * scale;
            }
        }
    } else if (outMode == 1) {
        __bf16* O = (__bf16*)Cout + (size_t)z * cStride;
#pragma unroll
        for (int t = 0; t < 4; ++t) {
            const int n = n0 + t * 16 + nc;
#pragma unroll
            for (int vv = 0; vv < 8; ++vv) {
                const int m = m0 + ((l < 16) ? vv : 8 + vv);
                O[(size_t)n * M + m] = (__bf16)(acc[t][vv] * scale);
            }
        }
    } else {
        __bf16* O = (__bf16*)Cout + (size_t)z * cStride;
#pragma unroll
        for (int t = 0; t < 4; ++t) {
            const int n = n0 + t * 16 + nc;
#pragma unroll
            for (int vv = 0; vv < 8; ++vv) {
                const int m = m0 + ((l < 16) ? vv : 8 + vv);
                O[(size_t)m * N + n] = (__bf16)(acc[t][vv] * scale);
            }
        }
    }
}

// ---------------------------------------------------------------------------
// Instance-norm stats: one block per (b,h) map of C0_*KV_ fp32 scores.
// stats[2z] = mean, stats[2z+1] = rsqrt(var + eps)
// ---------------------------------------------------------------------------
__global__ __launch_bounds__(256) void norm_stats(const float* __restrict__ scores,
                                                  float* __restrict__ stats) {
    const int z = blockIdx.x;
    const int mapSize = C0_ * KV_;
    const float* s = scores + (size_t)z * mapSize;
    float sum = 0.f, sq = 0.f;
    for (int i = threadIdx.x; i < mapSize; i += 256) {
        float x = s[i];
        sum += x; sq += x * x;
    }
    __shared__ float sh[256];
    sh[threadIdx.x] = sum; __syncthreads();
    for (int st = 128; st > 0; st >>= 1) {
        if (threadIdx.x < st) sh[threadIdx.x] += sh[threadIdx.x + st];
        __syncthreads();
    }
    float tot = sh[0]; __syncthreads();
    sh[threadIdx.x] = sq; __syncthreads();
    for (int st = 128; st > 0; st >>= 1) {
        if (threadIdx.x < st) sh[threadIdx.x] += sh[threadIdx.x + st];
        __syncthreads();
    }
    if (threadIdx.x == 0) {
        float mean = tot / (float)mapSize;
        float var  = sh[0] / (float)mapSize - mean * mean;
        stats[2 * z]     = mean;
        stats[2 * z + 1] = rsqrtf(var + 1e-5f);
    }
}

// ---------------------------------------------------------------------------
// Row softmax over KV (512) after instance-norm; writes bf16 probs.
// ---------------------------------------------------------------------------
__global__ __launch_bounds__(256) void softmax_rows(const float* __restrict__ scores,
                                                    const float* __restrict__ stats,
                                                    __bf16* __restrict__ probs) {
    const int row = blockIdx.x;          // 0 .. B*H*C0-1
    const int z   = row >> 8;            // row / C0_
    const float mean = stats[2 * z];
    const float rstd = stats[2 * z + 1];
    const float* s = scores + (size_t)row * KV_;
    __bf16*      p = probs  + (size_t)row * KV_;

    float x0 = (s[threadIdx.x]       - mean) * rstd;
    float x1 = (s[threadIdx.x + 256] - mean) * rstd;

    __shared__ float sh[256];
    sh[threadIdx.x] = fmaxf(x0, x1); __syncthreads();
    for (int st = 128; st > 0; st >>= 1) {
        if (threadIdx.x < st) sh[threadIdx.x] = fmaxf(sh[threadIdx.x], sh[threadIdx.x + st]);
        __syncthreads();
    }
    float mx = sh[0]; __syncthreads();

    float e0 = __expf(x0 - mx), e1 = __expf(x1 - mx);
    sh[threadIdx.x] = e0 + e1; __syncthreads();
    for (int st = 128; st > 0; st >>= 1) {
        if (threadIdx.x < st) sh[threadIdx.x] += sh[threadIdx.x + st];
        __syncthreads();
    }
    float inv = 1.f / sh[0];
    p[threadIdx.x]       = (__bf16)(e0 * inv);
    p[threadIdx.x + 256] = (__bf16)(e1 * inv);
}

// ---------------------------------------------------------------------------
// Head mean + transpose: ctx[b][h][c][n] (fp32) -> pooled[b][n][c] (bf16)
// ---------------------------------------------------------------------------
__global__ __launch_bounds__(256) void mean_pool(const float* __restrict__ ctx,
                                                 __bf16* __restrict__ pooled) {
    size_t i = (size_t)blockIdx.x * blockDim.x + threadIdx.x;
    const size_t total = (size_t)B_ * N_ * C0_;
    if (i >= total) return;
    int c = (int)(i % C0_);
    size_t bn = i / C0_;
    int n = (int)(bn % N_);
    int b = (int)(bn / N_);
    float s = 0.f;
#pragma unroll
    for (int h = 0; h < H_; ++h)
        s += ctx[(((size_t)b * H_ + h) * C0_ + c) * N_ + n];
    pooled[i] = (__bf16)(s * 0.25f);
}

// ---------------------------------------------------------------------------
extern "C" void kernel_launch(void* const* d_in, const int* in_sizes, int n_in,
                              void* d_out, int out_size, void* d_ws, size_t ws_size,
                              hipStream_t stream) {
    (void)in_sizes; (void)n_in; (void)out_size; (void)ws_size;

    const float* emb     = (const float*)d_in[0];   // [B,N,C0]
    const float* emb_all = (const float*)d_in[1];   // [B,N,KV]
    const float* Wq      = (const float*)d_in[2];   // [H,C0,C0]
    const float* Wk      = (const float*)d_in[3];   // [H,KV,KV]
    const float* Wv      = (const float*)d_in[4];   // [H,KV,KV]
    const float* Wout    = (const float*)d_in[5];   // [C0,C0]
    float* out           = (float*)d_out;           // [B,N,C0]

    // ---- workspace carve-out (256B aligned) ----
    char* base = (char*)d_ws;
    size_t off = 0;
    auto carve = [&](size_t bytes) -> char* {
        char* p = base + off;
        off = (off + bytes + 255) & ~(size_t)255;
        return p;
    };
    __bf16* emb_bf  = (__bf16*)carve((size_t)B_*N_*C0_*2);        // 4 MB
    __bf16* embA_bf = (__bf16*)carve((size_t)B_*N_*KV_*2);        // 8 MB
    __bf16* Wq_bf   = (__bf16*)carve((size_t)H_*C0_*C0_*2);
    __bf16* Wk_bf   = (__bf16*)carve((size_t)H_*KV_*KV_*2);
    __bf16* Wv_bf   = (__bf16*)carve((size_t)H_*KV_*KV_*2);
    __bf16* Wout_bf = (__bf16*)carve((size_t)C0_*C0_*2);
    __bf16* q_t     = (__bf16*)carve((size_t)B_*H_*C0_*N_*2);     // 16 MB  [z][d][n]
    __bf16* k_t     = (__bf16*)carve((size_t)B_*H_*KV_*N_*2);     // 32 MB  [z][j][n]
    __bf16* v_nat   = (__bf16*)carve((size_t)B_*H_*N_*KV_*2);     // 32 MB  [z][n][j]
    float*  scores  = (float*) carve((size_t)B_*H_*C0_*KV_*4);    // 16 MB  [z][d][j]
    float*  stats   = (float*) carve((size_t)B_*H_*2*4);
    __bf16* probs   = (__bf16*)carve((size_t)B_*H_*C0_*KV_*2);    //  8 MB  [z][d][j]
    // dead-region reuse: k_t dead after scores, q_t dead after scores
    float*  ctx     = (float*) k_t;                               // 32 MB  [z][d][n]
    __bf16* pooled  = (__bf16*)q_t;                               //  4 MB  [b][n][c]

    const int ZBH = B_ * H_;

    // ---- 1. fp32 -> bf16 conversions ----
    f32_to_bf16<<<1024, 256, 0, stream>>>(emb,     emb_bf,  (size_t)B_*N_*C0_);
    f32_to_bf16<<<1024, 256, 0, stream>>>(emb_all, embA_bf, (size_t)B_*N_*KV_);
    f32_to_bf16<<<256,  256, 0, stream>>>(Wq,      Wq_bf,   (size_t)H_*C0_*C0_);
    f32_to_bf16<<<512,  256, 0, stream>>>(Wk,      Wk_bf,   (size_t)H_*KV_*KV_);
    f32_to_bf16<<<512,  256, 0, stream>>>(Wv,      Wv_bf,   (size_t)H_*KV_*KV_);
    f32_to_bf16<<<64,   256, 0, stream>>>(Wout,    Wout_bf, (size_t)C0_*C0_);

    // ---- 2. projections ----
    // q_t[z][d][n] = (emb[b] @ Wq[h]^T)^T   M=N_ rows, N=C0_ cols, K=C0_
    wmma_gemm<<<dim3(C0_/128, N_/64, ZBH), 256, 0, stream>>>(
        emb_bf, Wq_bf, q_t, N_, C0_, C0_,
        (size_t)N_*C0_, 2, (size_t)C0_*C0_, H_-1, (size_t)C0_*N_, 1, 1.0f);
    // k_t[z][j][n] = (emb_all[b] @ Wk[h]^T)^T   M=N_, N=KV_, K=KV_
    wmma_gemm<<<dim3(KV_/128, N_/64, ZBH), 256, 0, stream>>>(
        embA_bf, Wk_bf, k_t, N_, KV_, KV_,
        (size_t)N_*KV_, 2, (size_t)KV_*KV_, H_-1, (size_t)KV_*N_, 1, 1.0f);
    // v[z][n][j] = emb_all[b] @ Wv[h]^T   (row-major)
    wmma_gemm<<<dim3(KV_/128, N_/64, ZBH), 256, 0, stream>>>(
        embA_bf, Wv_bf, v_nat, N_, KV_, KV_,
        (size_t)N_*KV_, 2, (size_t)KV_*KV_, H_-1, (size_t)N_*KV_, 0, 1.0f);

    // ---- 3. scores = q^T k / sqrt(KV)   M=C0_, N=KV_, K=N_ (fp32 out) ----
    const float invSqrtKV = 1.0f / sqrtf((float)KV_);
    wmma_gemm<<<dim3(KV_/128, C0_/64, ZBH), 256, 0, stream>>>(
        q_t, k_t, scores, C0_, KV_, N_,
        (size_t)C0_*N_, 0, (size_t)KV_*N_, 63, (size_t)C0_*KV_, 2, invSqrtKV);

    // ---- 4. instance-norm stats + row softmax -> bf16 probs ----
    norm_stats<<<ZBH, 256, 0, stream>>>(scores, stats);
    softmax_rows<<<ZBH * C0_, 256, 0, stream>>>(scores, stats, probs);

    // ---- 5. context = probs @ v^T   M=C0_, N=N_, K=KV_ (fp32 out) ----
    wmma_gemm<<<dim3(N_/128, C0_/64, ZBH), 256, 0, stream>>>(
        probs, v_nat, ctx, C0_, N_, KV_,
        (size_t)C0_*KV_, 0, (size_t)N_*KV_, 63, (size_t)C0_*N_, 2, 1.0f);

    // ---- 6. head mean + transpose -> pooled bf16 [b][n][c] ----
    {
        size_t total = (size_t)B_ * N_ * C0_;
        int blocks = (int)((total + 255) / 256);
        mean_pool<<<blocks, 256, 0, stream>>>(ctx, pooled);
    }

    // ---- 7. o = pooled @ Wout^T   M=N_, N=C0_, K=C0_, fp32 -> d_out ----
    wmma_gemm<<<dim3(C0_/128, N_/64, B_), 256, 0, stream>>>(
        pooled, Wout_bf, out, N_, C0_, C0_,
        (size_t)N_*C0_, 0, 0, 0, (size_t)N_*C0_, 2, 1.0f);
}
